// LinformerAttention_42588895707685
// MI455X (gfx1250) — matrix operand order repaired and checked by reading, not compiled, not run on hardware
//
#include <hip/hip_runtime.h>

// ---------------------------------------------------------------------------
// Linformer attention on gfx1250 (MI455X), bf16 WMMA pipeline, f32 accumulate.
// All matrix math uses v_wmma_f32_16x16x32_bf16 (probe-confirmed codegen).
// Native (__bf16) casts for f32->bf16 (hardware v_cvt, RNE).
// ---------------------------------------------------------------------------

typedef __attribute__((ext_vector_type(16))) __bf16 v16bf;
typedef __attribute__((ext_vector_type(8)))  __bf16 v8bf;
typedef __attribute__((ext_vector_type(8)))  float  v8f;

union B16x16 { v8bf h[2]; v16bf w; };

static constexpr int BATCH = 8;
static constexpr int SEQ   = 16384;
static constexpr int CDIM  = 48;     // input/output channel dim
static constexpr int HEADS = 3;
static constexpr int DH    = 16;
static constexpr int KLR   = 240;    // low-rank length
static constexpr int INNER = 48;     // HEADS*DH

static __device__ __forceinline__ v8f wmma_bf16(v16bf a, v16bf b, v8f c) {
    return __builtin_amdgcn_wmma_f32_16x16x32_bf16(false, a, false, b,
                                                   (short)0, c, false, false);
}

// --------------------------- kernel 0: f32 -> bf16 --------------------------
__global__ __launch_bounds__(256) void cvt_bf16_kernel(const float* __restrict__ src,
                                                       __bf16* __restrict__ dst,
                                                       int n) {
    int i = blockIdx.x * 256 + threadIdx.x;
    if (i < n) dst[i] = (__bf16)src[i];
}

// --------------------------- kernel 1: QKV projection -----------------------
// qkv[bn,e] = sum_c x[bn,c] * Wqkv[e,c];  e = h*48 + r, r<16 q, <32 k, else v.
// Q  : [B*H, SEQ, DH]    (A-layout friendly for dots GEMM)
// Kt : [B*H, DH, SEQ]    (B-layout friendly: K-dim = n contiguous)
// Vt : [B*H, DH, SEQ]
__global__ __launch_bounds__(256) void qkv_kernel(const float* __restrict__ x,
                                                  const __bf16* __restrict__ Wqkvbf,
                                                  __bf16* __restrict__ Q,
                                                  __bf16* __restrict__ Kt,
                                                  __bf16* __restrict__ Vt) {
    const int lane = threadIdx.x & 31;
    const int wave = threadIdx.x >> 5;
    const int lm = lane & 15, lh = lane >> 4;
    const int rowTile = blockIdx.x * 8 + wave;        // BN/16 = 8192 tiles
    const int row0 = rowTile * 16;

    // A fragments (rows = bn, K-dim = c, 48 padded to 64 in 2 chunks of 32)
    v16bf a0, a1 = {};
    {
        const float* xr = x + (size_t)(row0 + lm) * CDIM;
        v8f g0 = *(const v8f*)(xr + lh * 8);
        v8f g1 = *(const v8f*)(xr + 16 + lh * 8);
        v8f g2 = *(const v8f*)(xr + 32 + lh * 8);
        #pragma unroll
        for (int i = 0; i < 8; ++i) {
            a0[i]     = (__bf16)g0[i];
            a0[8 + i] = (__bf16)g1[i];
            a1[i]     = (__bf16)g2[i];     // c=48..63 stay 0
        }
    }

    #pragma unroll
    for (int et = 0; et < 9; ++et) {                  // 144/16 column tiles
        const int e = et * 16 + lm;
        const __bf16* wr = Wqkvbf + (size_t)e * CDIM; // B[c,e] = Wqkv[e,c]
        v8f c = {};
        {
            B16x16 bb; bb.h[0] = *(const v8bf*)(wr + lh * 8); bb.h[1] = *(const v8bf*)(wr + 16 + lh * 8); // c = lh*8+0..7, 16+lh*8+0..7
            c = wmma_bf16(a0, bb.w, c);
        }
        {
            B16x16 bb; v8bf z = {};
            bb.h[0] = *(const v8bf*)(wr + 32 + lh * 8); bb.h[1] = z;  // c = 32+lh*8+0..7
            c = wmma_bf16(a1, bb.w, c);
        }
        // scatter C (M = i + lh*8, N = lm)
        const int h = e / 48, r = e % 48;
        #pragma unroll
        for (int i = 0; i < 8; ++i) {
            const int bn = row0 + i + lh * 8;
            const int b = bn >> 14, n = bn & (SEQ - 1);
            const size_t bh = (size_t)(b * HEADS + h);
            const __bf16 val = (__bf16)c[i];
            if (r < 16)       Q [(bh * SEQ + n) * DH + r]        = val;
            else if (r < 32)  Kt[(bh * DH + (r - 16)) * SEQ + n] = val;
            else              Vt[(bh * DH + (r - 32)) * SEQ + n] = val;
        }
    }
}

// ------------------- kernel 2: low-rank sequence projection -----------------
// KP[b,h,k,d]  = sum_n E[h,k,n] * K[b,h,n,d]   (stored [B*H, KLR, DH] bf16)
// VPt[b,h,d,k] = sum_n F[h,k,n] * V[b,h,n,d]   (stored [B*H, DH, KLR] bf16)
__global__ __launch_bounds__(256) void proj_kernel(const __bf16* __restrict__ Ebf,
                                                   const __bf16* __restrict__ Fbf,
                                                   const __bf16* __restrict__ Kt,
                                                   const __bf16* __restrict__ Vt,
                                                   __bf16* __restrict__ KP,
                                                   __bf16* __restrict__ VPt) {
    __shared__ float red[8][2][32][8];                // 16 KB
    const int lane = threadIdx.x & 31;
    const int wave = threadIdx.x >> 5;
    const int lm = lane & 15, lh = lane >> 4;
    const int ktile = blockIdx.x % 15;
    const int bh = blockIdx.x / 15;
    const int h = bh % HEADS;
    const int krow = ktile * 16 + lm;

    const __bf16* eRow = Ebf + ((size_t)h * KLR + krow) * SEQ;
    const __bf16* fRow = Fbf + ((size_t)h * KLR + krow) * SEQ;
    const __bf16* kRow = Kt + ((size_t)bh * DH + lm) * SEQ;
    const __bf16* vRow = Vt + ((size_t)bh * DH + lm) * SEQ;

    v8f accK = {}, accV = {};
    const int n0 = wave * (SEQ / 8);
    for (int nc = n0; nc < n0 + (SEQ / 8); nc += 32) {
        B16x16 a; v16bf bb;
        a.h[0] = *(const v8bf*)(eRow + nc + lh * 16);
        a.h[1] = *(const v8bf*)(eRow + nc + lh * 16 + 8);
        bb = *(const v16bf*)(kRow + nc + lh * 16);
        accK = wmma_bf16(a.w, bb, accK);
        a.h[0] = *(const v8bf*)(fRow + nc + lh * 16);
        a.h[1] = *(const v8bf*)(fRow + nc + lh * 16 + 8);
        bb = *(const v16bf*)(vRow + nc + lh * 16);
        accV = wmma_bf16(a.w, bb, accV);
    }
    #pragma unroll
    for (int i = 0; i < 8; ++i) {
        red[wave][0][lane][i] = accK[i];
        red[wave][1][lane][i] = accV[i];
    }
    __syncthreads();
    if (wave < 2) {
        #pragma unroll
        for (int i = 0; i < 8; ++i) {
            float s = 0.f;
            #pragma unroll
            for (int wv = 0; wv < 8; ++wv) s += red[wv][wave][lane][i];
            const int k = ktile * 16 + i + lh * 8;
            const __bf16 val = (__bf16)s;
            if (wave == 0) KP [((size_t)bh * KLR + k) * DH + lm] = val;
            else           VPt[((size_t)bh * DH + lm) * KLR + k] = val;
        }
    }
}

// --------------------- kernel 3: attention core (per 16 rows) ---------------
__global__ __launch_bounds__(256) void attn_kernel(const __bf16* __restrict__ Q,
                                                   const __bf16* __restrict__ KP,
                                                   const __bf16* __restrict__ VPt,
                                                   __bf16* __restrict__ OutH) {
    __shared__ __bf16 attnLds[8][16][KLR];            // 60 KB, per-wave slabs
    const int lane = threadIdx.x & 31;
    const int wave = threadIdx.x >> 5;
    const int lm = lane & 15, lh = lane >> 4;
    const int bh = blockIdx.x / 128;                  // (b,h)
    const int nt = (blockIdx.x % 128) * 8 + wave;     // n-tile 0..1023
    const int n0 = nt * 16;

    // A = q rows (K-dim = d: 16 valid, upper half zero)
    v16bf aq;
    {
        B16x16 t;
        t.h[0] = *(const v8bf*)(Q + ((size_t)bh * SEQ + n0 + lm) * DH + lh * 8);
        v8bf z = {};
        t.h[1] = z;
        aq = t.w;
    }

    // dots[n,k] = q . KP[k,:]  (15 column tiles of 16)
    v8f dots[15];
    const v8f cz = {};
    #pragma unroll
    for (int t = 0; t < 15; ++t) {
        B16x16 u; v8bf z = {};
        u.h[0] = *(const v8bf*)(KP + ((size_t)bh * KLR + t * 16 + lm) * DH + lh * 8); u.h[1] = z;
        dots[t] = wmma_bf16(aq, u.w, cz);
    }

    // softmax over k (row = i + lh*8 spread across 16 lanes)
    #pragma unroll
    for (int t = 0; t < 15; ++t)
        #pragma unroll
        for (int i = 0; i < 8; ++i) dots[t][i] *= 0.25f;   // DH^-0.5
    #pragma unroll
    for (int i = 0; i < 8; ++i) {
        float m = dots[0][i];
        #pragma unroll
        for (int t = 1; t < 15; ++t) m = fmaxf(m, dots[t][i]);
        #pragma unroll
        for (int s = 1; s < 16; s <<= 1) m = fmaxf(m, __shfl_xor(m, s, 32));
        float sum = 0.f;
        #pragma unroll
        for (int t = 0; t < 15; ++t) { float e = __expf(dots[t][i] - m); dots[t][i] = e; sum += e; }
        #pragma unroll
        for (int s = 1; s < 16; s <<= 1) sum += __shfl_xor(sum, s, 32);
        const float inv = 1.0f / sum;
        #pragma unroll
        for (int t = 0; t < 15; ++t) dots[t][i] *= inv;
    }

    // stage attn (C-layout) into LDS as row-major bf16 for A-layout reload
    #pragma unroll
    for (int t = 0; t < 15; ++t)
        #pragma unroll
        for (int i = 0; i < 8; ++i)
            attnLds[wave][i + lh * 8][t * 16 + lm] = (__bf16)dots[t][i];
    __syncthreads();

    // out[n,d] = attn . VP  (K = 240, 8 chunks of 32; last chunk upper half 0)
    v8f outc = {};
    const __bf16* vp = VPt + (size_t)bh * DH * KLR + (size_t)lm * KLR;
    const __bf16* ar = &attnLds[wave][lm][0];
    #pragma unroll
    for (int kc = 0; kc < 256; kc += 32) {
        B16x16 a;
        a.h[0] = *(const v8bf*)(ar + kc + lh * 8);          // k <= 239 always
        if (kc < 224) a.h[1] = *(const v8bf*)(ar + kc + 16 + lh * 8);
        else {
            v8bf z = {};
            a.h[1] = z;                                     // k >= 240 -> 0
        }
        B16x16 bb; v8bf zb = {};
        bb.h[0] = *(const v8bf*)(vp + kc + lh * 8);
        bb.h[1] = (kc < 224) ? *(const v8bf*)(vp + kc + 16 + lh * 8) : zb;  // k >= 240 -> 0
        outc = wmma_bf16(a.w, bb.w, outc);
    }

    const int b = bh / HEADS, h = bh % HEADS;
    #pragma unroll
    for (int i = 0; i < 8; ++i) {
        const int n = n0 + i + lh * 8;
        OutH[((size_t)b * SEQ + n) * INNER + h * DH + lm] = (__bf16)outc[i];
    }
}

// --------------------------- kernel 4: output projection --------------------
// out[bn,d] = sum_e OutH[bn,e] * Wout[d,e] + bout[d]
__global__ __launch_bounds__(256) void outproj_kernel(const __bf16* __restrict__ OutH,
                                                      const __bf16* __restrict__ Woutbf,
                                                      const float* __restrict__ bout,
                                                      float* __restrict__ out) {
    const int lane = threadIdx.x & 31;
    const int wave = threadIdx.x >> 5;
    const int lm = lane & 15, lh = lane >> 4;
    const int rowTile = blockIdx.x * 8 + wave;
    const int row0 = rowTile * 16;

    v16bf a0, a1;
    {
        const __bf16* ar = OutH + (size_t)(row0 + lm) * INNER;
        B16x16 t;
        t.h[0] = *(const v8bf*)(ar + lh * 8);
        t.h[1] = *(const v8bf*)(ar + 16 + lh * 8);
        a0 = t.w;
        B16x16 t1;
        t1.h[0] = *(const v8bf*)(ar + 32 + lh * 8);
        v8bf z = {};
        t1.h[1] = z;                                        // e=48..63 -> 0
        a1 = t1.w;
    }
    #pragma unroll
    for (int dt = 0; dt < 3; ++dt) {
        const int d = dt * 16 + lm;
        const __bf16* wr = Woutbf + (size_t)d * CDIM;       // B[e,d]=Wout[d,e]
        v8f c = {};
        {
            B16x16 bb; bb.h[0] = *(const v8bf*)(wr + lh * 8); bb.h[1] = *(const v8bf*)(wr + 16 + lh * 8); // e = lh*8+0..7, 16+lh*8+0..7
            c = wmma_bf16(a0, bb.w, c);
        }
        {
            B16x16 bb; v8bf z = {};
            bb.h[0] = *(const v8bf*)(wr + 32 + lh * 8); bb.h[1] = z;     // e = 32+lh*8+0..7
            c = wmma_bf16(a1, bb.w, c);
        }
        const float bias = bout[d];
        #pragma unroll
        for (int i = 0; i < 8; ++i)
            out[(size_t)(row0 + i + lh * 8) * CDIM + d] = c[i] + bias;
    }
}

// ---------------------------------------------------------------------------
extern "C" void kernel_launch(void* const* d_in, const int* in_sizes, int n_in,
                              void* d_out, int out_size, void* d_ws, size_t ws_size,
                              hipStream_t stream) {
    const float* x    = (const float*)d_in[0];
    const float* Wqkv = (const float*)d_in[1];
    const float* E    = (const float*)d_in[2];
    const float* F    = (const float*)d_in[3];
    const float* Wout = (const float*)d_in[4];
    const float* bout = (const float*)d_in[5];
    float* out = (float*)d_out;

    // workspace carving (bf16 elements); total ~98 MB
    __bf16* ws = (__bf16*)d_ws;
    size_t o = 0;
    __bf16* Ebf    = ws + o; o += (size_t)HEADS * KLR * SEQ;
    __bf16* Fbf    = ws + o; o += (size_t)HEADS * KLR * SEQ;
    __bf16* Q      = ws + o; o += (size_t)BATCH * HEADS * SEQ * DH;
    __bf16* Kt     = ws + o; o += (size_t)BATCH * HEADS * SEQ * DH;
    __bf16* Vt     = ws + o; o += (size_t)BATCH * HEADS * SEQ * DH;
    __bf16* KP     = ws + o; o += (size_t)BATCH * HEADS * KLR * DH;
    __bf16* VPt    = ws + o; o += (size_t)BATCH * HEADS * KLR * DH;
    __bf16* OutH   = ws + o; o += (size_t)BATCH * SEQ * INNER;
    __bf16* Wqkvbf = ws + o; o += (size_t)3 * INNER * CDIM;
    __bf16* Woutbf = ws + o; o += (size_t)INNER * CDIM;

    const int nE = HEADS * KLR * SEQ;
    cvt_bf16_kernel<<<(nE + 255) / 256, 256, 0, stream>>>(E, Ebf, nE);
    cvt_bf16_kernel<<<(nE + 255) / 256, 256, 0, stream>>>(F, Fbf, nE);
    cvt_bf16_kernel<<<(3 * INNER * CDIM + 255) / 256, 256, 0, stream>>>(Wqkv, Wqkvbf, 3 * INNER * CDIM);
    cvt_bf16_kernel<<<(INNER * CDIM + 255) / 256, 256, 0, stream>>>(Wout, Woutbf, INNER * CDIM);

    qkv_kernel<<<(BATCH * SEQ / 16) / 8, 256, 0, stream>>>(x, Wqkvbf, Q, Kt, Vt);
    proj_kernel<<<BATCH * HEADS * (KLR / 16), 256, 0, stream>>>(Ebf, Fbf, Kt, Vt, KP, VPt);
    attn_kernel<<<BATCH * HEADS * (SEQ / 16 / 8), 256, 0, stream>>>(Q, KP, VPt, OutH);
    outproj_kernel<<<(BATCH * SEQ / 16) / 8, 256, 0, stream>>>(OutH, Woutbf, bout, out);

    (void)in_sizes; (void)n_in; (void)out_size; (void)ws_size;
}
